// BoundedGridLocNet_6047313952947
// MI455X (gfx1250) — compile-verified, matrix operand order checked by reading, not run on hardware
//
#include <hip/hip_runtime.h>
#include <hip/hip_bf16.h>

// ---------------------------------------------------------------------------
// Types for CDNA5 WMMA fragments (wave32, 16x16x32 bf16 -> f32) and TDM D#
// ---------------------------------------------------------------------------
typedef __attribute__((ext_vector_type(16))) __bf16         v16bf;
typedef __attribute__((ext_vector_type(8)))  float          v8f;
typedef __attribute__((ext_vector_type(16))) unsigned short v16u;
typedef __attribute__((ext_vector_type(8)))  unsigned short v8u;
typedef __attribute__((ext_vector_type(4)))  unsigned int   v4u32;
typedef __attribute__((ext_vector_type(8)))  int            v8i32;
typedef __attribute__((ext_vector_type(4)))  int            v4i32;

__device__ __forceinline__ unsigned short f32_to_bf16_rne(float f) {
  unsigned int u = __float_as_uint(f);
  u += 0x7FFFu + ((u >> 16) & 1u);
  return (unsigned short)(u >> 16);
}

// ---------------------------------------------------------------------------
// TDM: stage a (rows x Kpad) bf16 weight tile (row-major, contiguous) into
// LDS. 2D descriptor per CDNA5 ISA. Wave 0 only.
// ---------------------------------------------------------------------------
__device__ __forceinline__ void tdm_load_wtile(const unsigned short* gsrc,
                                               unsigned short* lds,
                                               int Kpad, int rows) {
  const unsigned long long ga = (unsigned long long)(const void*)gsrc;
  const unsigned int ldsoff = (unsigned int)(unsigned long long)(void*)lds;
  v4u32 g0;
  g0[0] = 1u;                                   // count=1, user descriptor
  g0[1] = ldsoff;                               // lds_addr (bytes)
  g0[2] = (unsigned int)(ga & 0xFFFFFFFFu);     // global_addr[31:0]
  g0[3] = (unsigned int)((ga >> 32) & 0x01FFFFFFu) | (2u << 30); // [56:32]|type=2
  v8i32 g1;
  g1[0] = 0x10000;                              // wg_mask=0, data_size=2B
  g1[1] = (int)((unsigned int)Kpad << 16);      // tensor_dim0[15:0]
  g1[2] = (int)(((unsigned int)Kpad >> 16) | ((unsigned int)rows << 16));
  g1[3] = (int)((unsigned int)Kpad << 16);      // td1 hi=0 | tile_dim0=Kpad
  g1[4] = rows;                                 // tile_dim1=rows, tile_dim2=0
  g1[5] = Kpad;                                 // tensor_dim0_stride[31:0]
  g1[6] = 0;
  g1[7] = 0;
  v4i32 z = {0, 0, 0, 0};
#if defined(__clang_major__) && (__clang_major__ >= 23)
  v8i32 z8 = {0, 0, 0, 0, 0, 0, 0, 0};
  __builtin_amdgcn_tensor_load_to_lds(g0, g1, z, z, z8, 0);
#else
  __builtin_amdgcn_tensor_load_to_lds(g0, g1, z, z, 0);
#endif
  __builtin_amdgcn_s_wait_tensorcnt(0);
}

// ---------------------------------------------------------------------------
// Weight convert: f32 (Co x K) row-major -> bf16 (CoPad x Kpad) zero-padded
// ---------------------------------------------------------------------------
__global__ void wcvt_kernel(const float* __restrict__ w,
                            unsigned short* __restrict__ out,
                            int Co, int K, int CoPad, int Kpad) {
  int idx = blockIdx.x * blockDim.x + threadIdx.x;
  int total = CoPad * Kpad;
  if (idx >= total) return;
  int co = idx / Kpad;
  int k  = idx - co * Kpad;
  float v = (co < Co && k < K) ? w[(size_t)co * K + k] : 0.f;
  out[idx] = f32_to_bf16_rne(v);
}

// ---------------------------------------------------------------------------
// Implicit-GEMM conv (+bias) via v_wmma_f32_16x16x32_bf16, templated on
// kernel size / stride / N-tiles per wave. 256 threads = 8 waves; the
// block's NT*16-channel weight tile is TDM-staged into LDS once, B fragments
// come from LDS (ds_load_b128); each wave owns one 16-pixel M tile and NT
// accumulators (one A gather feeds NT WMMAs). A loads are branchless:
// address clamped + value zeroed by v_cndmask (no EXEC save/restore).
// ---------------------------------------------------------------------------
template <int KS, int STRIDE, int NT>
__global__ void conv_wmma_kernel(const float* __restrict__ src,
                                 const unsigned short* __restrict__ wbf,
                                 const float* __restrict__ bias,
                                 float* __restrict__ dst,
                                 int Ci, int Hi, int Wi,
                                 int Co, int Ho, int Wo, int Kpad) {
  extern __shared__ unsigned short smw[];     // (NT*16) * Kpad bf16 weights
  constexpr int KK  = KS * KS;
  constexpr int PAD = 1;

  const int tid   = threadIdx.x;
  const int lane  = tid & 31;
  const int wave  = tid >> 5;
  const int l15   = lane & 15;       // A row / B col / D col
  const int h     = lane >> 4;       // K-half (A/B), row-half (D)
  const int nbase = blockIdx.y * (16 * NT);
  const int b     = blockIdx.z;
  const int M     = Ho * Wo;

  if (wave == 0)
    tdm_load_wtile(wbf + (size_t)nbase * Kpad, smw, Kpad, 16 * NT);
  __syncthreads();

  const int mbase = blockIdx.x * 128 + wave * 16;
  if (mbase < M) {
    const int pixel = mbase + l15;
    const int mval  = pixel < M;
    const int oy    = pixel / Wo;
    const int ox    = pixel - oy * Wo;
    const int iy0   = oy * STRIDE - PAD;
    const int ix0   = ox * STRIDE - PAD;
    const float* sb = src + (size_t)b * Ci * Hi * Wi;

    v8f acc[NT];
#pragma unroll
    for (int t = 0; t < NT; ++t) acc[t] = (v8f){};

#pragma unroll 2
    for (int kb = 0; kb < Kpad; kb += 32) {
      v16bf a;
#pragma unroll
      for (int e = 0; e < 16; ++e) {
        const int vg = e >> 1, p = e & 1;
        const int kk = kb + ((vg < 4) ? (2 * vg + p + 8 * h)
                                      : (16 + 2 * (vg - 4) + p + 8 * h));
        const int ci = kk / KK;                // compile-time KK
        const int rr = kk - ci * KK;
        const int kh = rr / KS;
        const int kw = rr - kh * KS;
        const int iy = iy0 + kh;
        const int ix = ix0 + kw;
        const int ok = mval & (ci < Ci) & (iy >= 0) & (iy < Hi) &
                       (ix >= 0) & (ix < Wi);
        const int idx = (ci * Hi + iy) * Wi + ix;
        float v = sb[ok ? idx : 0];            // unconditional load
        a[e] = (__bf16)(ok ? v : 0.f);         // native f32->bf16 cvt
      }
#pragma unroll
      for (int t = 0; t < NT; ++t) {
        const unsigned short* wr =
            smw + (size_t)(t * 16 + l15) * Kpad + h * 8;
        v16u bu;
        ((v8u*)&bu)[0] = *(const v8u*)(wr + kb);        // ds_load_b128
        ((v8u*)&bu)[1] = *(const v8u*)(wr + kb + 16);
        acc[t] = __builtin_amdgcn_wmma_f32_16x16x32_bf16(
            false, a, false, __builtin_bit_cast(v16bf, bu),
            (short)0, acc[t], false, false);
      }
    }

#pragma unroll
    for (int t = 0; t < NT; ++t) {
      const int co = nbase + t * 16 + l15;
      if (co < Co) {
        const float bv = bias[co];
        float* db = dst + ((size_t)b * Co + co) * (size_t)M;
#pragma unroll
        for (int r = 0; r < 8; ++r) {
          const int pm = mbase + h * 8 + r;
          if (pm < M) db[pm] = acc[t][r] + bv;
        }
      }
    }
  }
}

// ---------------------------------------------------------------------------
// Dense GEMM for the FC layers: out[M x N] = A[M x K](f32) * B^T + bias.
// B pre-padded bf16 (NPad x K). actmode 0 = ReLU, 1 = tanh.
// ---------------------------------------------------------------------------
__global__ void gemm_wmma_kernel(const float* __restrict__ A,
                                 const unsigned short* __restrict__ Bw,
                                 const float* __restrict__ bias,
                                 float* __restrict__ out,
                                 int Mrows, int K, int Nactual, int ldOut,
                                 int actmode) {
  const int lane  = threadIdx.x;
  const int l15   = lane & 15;
  const int h     = lane >> 4;
  const int mbase = blockIdx.x * 16;
  const int nbase = blockIdx.y * 16;

  const int  m    = mbase + l15;
  const int  mval = m < Mrows;
  const float* arow = A + (size_t)(mval ? m : 0) * K;
  const unsigned short* wrow = Bw + (size_t)(nbase + l15) * K + h * 8;

  v8f acc = {};
  for (int kb = 0; kb < K; kb += 32) {
    __builtin_prefetch(wrow + kb + 32, 0, 3);   // global_prefetch_b8 (near)
    v16bf a;
#pragma unroll
    for (int e = 0; e < 16; ++e) {
      const int vg = e >> 1, p = e & 1;
      const int kk = kb + ((vg < 4) ? (2 * vg + p + 8 * h)
                                    : (16 + 2 * (vg - 4) + p + 8 * h));
      float v = arow[kk];                       // row clamped, always safe
      a[e] = (__bf16)(mval ? v : 0.f);
    }
    v16u bu;
    ((v8u*)&bu)[0] = *(const v8u*)(wrow + kb);
    ((v8u*)&bu)[1] = *(const v8u*)(wrow + kb + 16);
    acc = __builtin_amdgcn_wmma_f32_16x16x32_bf16(
        false, a, false, __builtin_bit_cast(v16bf, bu),
        (short)0, acc, false, false);
  }

  const int n = nbase + l15;
  if (n < Nactual) {
    const float bv = bias[n];
#pragma unroll
    for (int r = 0; r < 8; ++r) {
      const int mm = mbase + h * 8 + r;
      if (mm < Mrows) {
        float v = acc[r] + bv;
        v = actmode ? tanhf(v) : fmaxf(v, 0.f);
        out[(size_t)mm * ldOut + n] = v;
      }
    }
  }
}

// ---------------------------------------------------------------------------
// Instance norm, in place on one (b,c) plane of HW f32 elements.
// preRelu: relu before statistics (layer 0); postRelu: relu after norm.
// ---------------------------------------------------------------------------
__global__ void inorm_kernel(float* __restrict__ buf, int HW,
                             int preRelu, int postRelu) {
  __shared__ float s1[256];
  __shared__ float s2[256];
  float* p = buf + (size_t)blockIdx.x * HW;
  const int tid = threadIdx.x;

  float a = 0.f, q = 0.f;
  for (int i = tid; i < HW; i += 256) {
    float v = p[i];
    if (preRelu) v = fmaxf(v, 0.f);
    a += v;
    q += v * v;
  }
  s1[tid] = a; s2[tid] = q;
  __syncthreads();
  for (int st = 128; st > 0; st >>= 1) {
    if (tid < st) { s1[tid] += s1[tid + st]; s2[tid] += s2[tid + st]; }
    __syncthreads();
  }
  const float invHW = 1.f / (float)HW;
  const float mean  = s1[0] * invHW;
  const float var   = s2[0] * invHW - mean * mean;
  const float rs    = rsqrtf(var + 1e-5f);

  for (int i = tid; i < HW; i += 256) {
    float v = p[i];
    if (preRelu) v = fmaxf(v, 0.f);
    float o = (v - mean) * rs;
    if (postRelu) o = fmaxf(o, 0.f);
    p[i] = o;
  }
}

// ---------------------------------------------------------------------------
// Max pool (1,1,2,2)/(1,1,2,2) VALID on (64,256,4,3) -> flattened (64,512)
// ---------------------------------------------------------------------------
__global__ void pool_kernel(const float* __restrict__ in,
                            float* __restrict__ out) {
  int i = blockIdx.x * blockDim.x + threadIdx.x;
  if (i >= 64 * 512) return;
  int b   = i >> 9;
  int rem = i & 511;
  int c   = rem >> 1;
  int oy  = rem & 1;
  const float* p = in + (((size_t)b * 256 + c) * 4 + 2 * oy) * 3;
  out[i] = fmaxf(fmaxf(p[0], p[1]), fmaxf(p[3], p[4]));  // ix=2 dropped
}

// ---------------------------------------------------------------------------
// Grid-regularity losses on coor (64 x 25 x 2) -> 6 scalars
// ---------------------------------------------------------------------------
__global__ void loss_kernel(const float* __restrict__ coor,
                            float* __restrict__ outs) {
  __shared__ float srx[64], sry[64], scx[64], scy[64];
  const int b = threadIdx.x;  // blockDim == 64
  const float* g = coor + b * 50;  // [5][5][2]
  float rx = 0.f, ry = 0.f, cx = 0.f, cy = 0.f;
  for (int i = 0; i < 5; ++i) {
    for (int j = 0; j < 3; ++j) {
      float d1x = g[(i*5+j+1)*2]   - g[(i*5+j)*2];     d1x *= d1x;
      float d2x = g[(i*5+j+2)*2]   - g[(i*5+j+1)*2];   d2x *= d2x;
      float d1y = g[(i*5+j+1)*2+1] - g[(i*5+j)*2+1];   d1y *= d1y;
      float d2y = g[(i*5+j+2)*2+1] - g[(i*5+j+1)*2+1]; d2y *= d2y;
      rx += fmaxf(0.08f, fabsf(d2x - d1x));
      ry += fmaxf(0.08f, fabsf(d2y - d1y));
      float e1x = g[((j+1)*5+i)*2]   - g[(j*5+i)*2];     e1x *= e1x;
      float e2x = g[((j+2)*5+i)*2]   - g[((j+1)*5+i)*2]; e2x *= e2x;
      float e1y = g[((j+1)*5+i)*2+1] - g[(j*5+i)*2+1];   e1y *= e1y;
      float e2y = g[((j+2)*5+i)*2+1] - g[((j+1)*5+i)*2+1]; e2y *= e2y;
      cx += fmaxf(0.08f, fabsf(e2x - e1x));
      cy += fmaxf(0.08f, fabsf(e2y - e1y));
    }
  }
  srx[b] = rx; sry[b] = ry; scx[b] = cx; scy[b] = cy;
  __syncthreads();
  if (b == 0) {
    float RX = 0.f, RY = 0.f, CX = 0.f, CY = 0.f;
    for (int i = 0; i < 64; ++i) {
      RX += srx[i]; RY += sry[i]; CX += scx[i]; CY += scy[i];
    }
    const float inv = 1.f / (64.f * 15.f);
    const float* g0 = coor;
    float rg = 0.f, cg = 0.f;
    for (int r = 0; r < 5; ++r) {
      for (int t = 0; t < 3; ++t) {
        float x0 = g0[(r*5+t)*2],   y0 = g0[(r*5+t)*2+1];
        float x1 = g0[(r*5+t+1)*2], y1 = g0[(r*5+t+1)*2+1];
        float x2 = g0[(r*5+t+2)*2], y2 = g0[(r*5+t+2)*2+1];
        rg += fabsf((y1 - y0) * (x1 - x2) - (y1 - y2) * (x1 - x0));
        float u0 = g0[(t*5+r)*2],     v0 = g0[(t*5+r)*2+1];
        float u1 = g0[((t+1)*5+r)*2], v1 = g0[((t+1)*5+r)*2+1];
        float u2 = g0[((t+2)*5+r)*2], v2 = g0[((t+2)*5+r)*2+1];
        cg += fabsf((v1 - v0) * (u1 - u2) - (v1 - v2) * (u1 - u0));
      }
    }
    outs[0] = RX * inv;
    outs[1] = RY * inv;
    outs[2] = CX * inv;
    outs[3] = CY * inv;
    outs[4] = fmaxf(rg, 0.02f);
    outs[5] = fmaxf(cg, 0.02f);
  }
}

// ---------------------------------------------------------------------------
// Host side
// ---------------------------------------------------------------------------
extern "C" void kernel_launch(void* const* d_in, const int* in_sizes, int n_in,
                              void* d_out, int out_size, void* d_ws, size_t ws_size,
                              hipStream_t stream) {
  (void)in_sizes; (void)n_in; (void)out_size; (void)ws_size;
  const int B = 64;
  const int CiA[8] = {  5,   8,  16,  32,  64, 128, 256, 256};
  const int CoA[8] = {  8,  16,  32,  64, 128, 256, 256, 256};
  const int HiA[8] = {256, 128,  64,  32,  16,   8,   4,   4};
  const int WiA[8] = {192,  96,  48,  24,  12,   6,   3,   3};
  const int HoA[8] = {128,  64,  32,  16,   8,   4,   4,   4};
  const int WoA[8] = { 96,  48,  24,  12,   6,   3,   3,   3};
  const int KsA[8] = {  4,   4,   4,   4,   4,   4,   3,   3};

  int KA[8], KpadA[8], CoPadA[8];
  for (int l = 0; l < 8; ++l) {
    KA[l]    = CiA[l] * KsA[l] * KsA[l];
    KpadA[l] = (KA[l] + 31) & ~31;
    CoPadA[l]= (CoA[l] + 15) & ~15;
  }

  // workspace layout (256-byte aligned slabs)
  char* ws = (char*)d_ws;
  size_t off = 0;
  auto alloc = [&](size_t bytes) -> size_t {
    off = (off + 255) & ~(size_t)255;
    size_t o = off; off += bytes; return o;
  };
  size_t actOff[8], wOff[8];
  for (int l = 0; l < 8; ++l)
    actOff[l] = alloc((size_t)B * CoA[l] * HoA[l] * WoA[l] * sizeof(float));
  for (int l = 0; l < 8; ++l)
    wOff[l] = alloc((size_t)CoPadA[l] * KpadA[l] * sizeof(unsigned short));
  size_t fc1wOff = alloc((size_t)128 * 512 * sizeof(unsigned short));
  size_t fc2wOff = alloc((size_t)64  * 128 * sizeof(unsigned short));
  size_t poolOff = alloc((size_t)B * 512 * sizeof(float));
  size_t fc1aOff = alloc((size_t)B * 128 * sizeof(float));

  // inputs: x, (w0,b0)..(w7,b7), fc1_w, fc1_b, fc2_w, fc2_b
  const float* x = (const float*)d_in[0];
  const float* wP[8]; const float* bP[8];
  for (int l = 0; l < 8; ++l) {
    wP[l] = (const float*)d_in[1 + 2 * l];
    bP[l] = (const float*)d_in[2 + 2 * l];
  }
  const float* fc1w = (const float*)d_in[17];
  const float* fc1b = (const float*)d_in[18];
  const float* fc2w = (const float*)d_in[19];
  const float* fc2b = (const float*)d_in[20];
  float* out = (float*)d_out;

  // 1. convert + pad all weights to bf16
  for (int l = 0; l < 8; ++l) {
    int tot = CoPadA[l] * KpadA[l];
    wcvt_kernel<<<(tot + 255) / 256, 256, 0, stream>>>(
        wP[l], (unsigned short*)(ws + wOff[l]),
        CoA[l], KA[l], CoPadA[l], KpadA[l]);
  }
  wcvt_kernel<<<(128 * 512 + 255) / 256, 256, 0, stream>>>(
      fc1w, (unsigned short*)(ws + fc1wOff), 128, 512, 128, 512);
  wcvt_kernel<<<(64 * 128 + 255) / 256, 256, 0, stream>>>(
      fc2w, (unsigned short*)(ws + fc2wOff), 50, 128, 64, 128);

  // 2. conv stack, TDM-staged weights, per-layer instance norm
  for (int l = 0; l < 8; ++l) {
    const float* src = (l == 0) ? x : (const float*)(ws + actOff[l - 1]);
    float* dst = (float*)(ws + actOff[l]);
    int M  = HoA[l] * WoA[l];
    int NT = (CoPadA[l] >= 32) ? 2 : 1;
    dim3 grid((M + 127) / 128, CoPadA[l] / (16 * NT), B);
    size_t shmem = (size_t)16 * NT * KpadA[l] * sizeof(unsigned short);
    const unsigned short* wbf = (const unsigned short*)(ws + wOff[l]);
    if (KsA[l] == 4 && NT == 1) {
      conv_wmma_kernel<4, 2, 1><<<grid, 256, shmem, stream>>>(
          src, wbf, bP[l], dst,
          CiA[l], HiA[l], WiA[l], CoA[l], HoA[l], WoA[l], KpadA[l]);
    } else if (KsA[l] == 4) {
      conv_wmma_kernel<4, 2, 2><<<grid, 256, shmem, stream>>>(
          src, wbf, bP[l], dst,
          CiA[l], HiA[l], WiA[l], CoA[l], HoA[l], WoA[l], KpadA[l]);
    } else {
      conv_wmma_kernel<3, 1, 2><<<grid, 256, shmem, stream>>>(
          src, wbf, bP[l], dst,
          CiA[l], HiA[l], WiA[l], CoA[l], HoA[l], WoA[l], KpadA[l]);
    }
    // layer 0: inorm(relu(conv));  layers 1..7: relu(inorm(conv))
    inorm_kernel<<<B * CoA[l], 256, 0, stream>>>(
        dst, M, (l == 0) ? 1 : 0, (l == 0) ? 0 : 1);
  }

  // 3. maxpool -> (64, 512)
  pool_kernel<<<(64 * 512 + 255) / 256, 256, 0, stream>>>(
      (const float*)(ws + actOff[7]), (float*)(ws + poolOff));

  // 4. fc1 (relu) and fc2 (tanh -> coor in d_out[0:3200])
  gemm_wmma_kernel<<<dim3(4, 8), 32, 0, stream>>>(
      (const float*)(ws + poolOff), (const unsigned short*)(ws + fc1wOff),
      fc1b, (float*)(ws + fc1aOff), 64, 512, 128, 128, 0);
  gemm_wmma_kernel<<<dim3(4, 4), 32, 0, stream>>>(
      (const float*)(ws + fc1aOff), (const unsigned short*)(ws + fc2wOff),
      fc2b, out, 64, 128, 50, 50, 1);

  // 5. regularity losses -> d_out[3200:3206]
  loss_kernel<<<1, 64, 0, stream>>>(out, out + 3200);
}
